// LearnableSTFT_17308718203485
// MI455X (gfx1250) — matrix operand order repaired and checked by reading, not compiled, not run on hardware
//
#include <hip/hip_runtime.h>

typedef __attribute__((ext_vector_type(16))) _Float16 v16h;
typedef __attribute__((ext_vector_type(8)))  _Float16 v8h;
typedef __attribute__((ext_vector_type(2)))  _Float16 v2h;
typedef __attribute__((ext_vector_type(8)))  float    v8f;

// Problem constants (from setup_inputs): B=32, N=262144, win=hop-framed STFT
#define DFT      1024
#define HOP      256
#define BATCH    32
#define NSIG     262144
#define NFRAMES  1021          // ceil((262144-1024)/256)+1, zero pad
#define KSTEP    32            // K per v_wmma_f32_16x16x32_f16
#define NKC      (DFT / KSTEP) // 32 K-chunks
#define NTILES_N (DFT / 16)    // 64 bin tiles of 16
#define WAVES    8
#define BLK_BINS (WAVES * 16)  // 128 bins per block
#define FR_TILE  32            // 32 frames per block: 2 M-tiles per wave
#define A_PITCH  40            // halves per LDS row: 80B (16B-aligned, bank-skewed)

// ---------------------------------------------------------------------------
// Kernel 1: bake DFT twiddles into workspace in WMMA B-fragment order,
// split into f16 hi/lo pairs (Markidis precision splitting).
// B-fragment order chosen here (16-bit dense 32x16, same k order as A): lane n(0..15) holds col N=n,
// K contiguous (0..15 / 16..31 by lane half).
// Storage: frag[kc][ntile][lane][e] -> per-lane 32B contiguous & coalesced.
// ---------------------------------------------------------------------------
__global__ void dft_pack_kernel(_Float16* __restrict__ ch, _Float16* __restrict__ cl,
                                _Float16* __restrict__ sh, _Float16* __restrict__ sl) {
    int t = blockIdx.x * blockDim.x + threadIdx.x;   // 0 .. 1024*1024-1
    int k = t >> 10;                                  // DFT bin
    int n = t & 1023;                                 // window sample
    int m = (k * n) & 1023;                           // exact angle reduction
    float ang = -6.2831853071795864f * (1.0f / 1024.0f) * (float)m;
    float s, c;
    __sincosf(ang, &s, &c);

    int kc   = k >> 5;
    int kk   = k & 31;
    int lane = ((kk >> 4) << 4) | (n & 15);
    int e    = kk & 15;
    size_t idx = (((size_t)(kc * NTILES_N + (n >> 4)) * 32 + lane) << 4) + e;

    _Float16 chh = (_Float16)c;
    ch[idx] = chh;
    cl[idx] = (_Float16)(c - (float)chh);
    _Float16 shh = (_Float16)s;
    sh[idx] = shh;
    sl[idx] = (_Float16)(s - (float)shh);
}

// ---------------------------------------------------------------------------
// Kernel 2: STFT as split-precision f16 WMMA GEMM.
// Block = 256 threads (8 wave32), tile = 32 frames x 128 bins, re+im together.
// Per K-step per wave: 12x v_wmma_f32_16x16x32_f16 (2 M-tiles x 3-split x
// {cos,sin}) against a single set of 4 B-fragment loads.
// ---------------------------------------------------------------------------
__global__ __launch_bounds__(256)
void stft_wmma_kernel(const float* __restrict__ sig, const float* __restrict__ win,
                      const _Float16* __restrict__ ch, const _Float16* __restrict__ cl,
                      const _Float16* __restrict__ sh, const _Float16* __restrict__ sl,
                      float* __restrict__ out) {
    __shared__ float    Wlds[DFT];
    __shared__ _Float16 AsH[FR_TILE * A_PITCH];
    __shared__ _Float16 AsL[FR_TILE * A_PITCH];

    const int tid    = threadIdx.x;
    const int lane   = tid & 31;
    const int laneHi = lane >> 4;
    const int wave   = tid >> 5;
    const int b      = blockIdx.z;
    const int f0     = blockIdx.y * FR_TILE;            // first frame of tile
    const int ntg    = blockIdx.x * WAVES + wave;       // global 16-bin tile id

    // ---- preload window into LDS once ----
    {
        const float2* w2  = (const float2*)win;
        float2*       wl2 = (float2*)Wlds;
        wl2[tid]       = w2[tid];
        wl2[tid + 256] = w2[tid + 256];
    }
    __syncthreads();

    const float* sigB = sig + (size_t)b * NSIG;

    v8f accRe0 = {}, accIm0 = {}, accRe1 = {}, accIm1 = {};

    // staging coords: 32 rows x 16 float2-pairs, 256 threads x 2 iters
    const int rq_r = tid >> 4;   // 0..15 (+16 on second iter)
    const int rq_q = tid & 15;   // pair index within K-chunk

    for (int kc = 0; kc < NKC; ++kc) {
        const int kg = kc * KSTEP + rq_q * 2;

        // ---- stage A: branchless windowed f32 -> f16 hi/lo split ----
        #pragma unroll
        for (int it = 0; it < 2; ++it) {
            int   r  = rq_r + it * 16;
            int   f  = f0 + r;
            int   fc = (f < NFRAMES) ? f : (NFRAMES - 1);   // clamp: load always legal
            float2 xv = *(const float2*)(sigB + (size_t)fc * HOP + kg);
            float2 wv = *(const float2*)(Wlds + kg);
            float x0 = (f < NFRAMES) ? xv.x * wv.x : 0.0f;  // v_cndmask, no branch
            float x1 = (f < NFRAMES) ? xv.y * wv.y : 0.0f;
            _Float16 h0 = (_Float16)x0;
            _Float16 h1 = (_Float16)x1;
            v2h hh = {h0, h1};
            v2h ll = {(_Float16)(x0 - (float)h0), (_Float16)(x1 - (float)h1)};
            *(v2h*)(&AsH[r * A_PITCH + rq_q * 2]) = hh;
            *(v2h*)(&AsL[r * A_PITCH + rq_q * 2]) = ll;
        }
        __syncthreads();

        // ---- A fragments from LDS (contiguous K = 16*laneHi + i, same order as B packing), 2 M-tiles ----
        v16h aH0, aL0, aH1, aL1;
        #pragma unroll
        for (int mt = 0; mt < 2; ++mt) {
            int row = mt * 16 + (lane & 15);
            const _Float16* bH = &AsH[row * A_PITCH + laneHi * 16];
            const _Float16* bL = &AsL[row * A_PITCH + laneHi * 16];
            v8h h0 = *(const v8h*)(bH);
            v8h h1 = *(const v8h*)(bH + 8);
            v8h l0 = *(const v8h*)(bL);
            v8h l1 = *(const v8h*)(bL + 8);
            v16h aH, aL;
            #pragma unroll
            for (int i = 0; i < 8; ++i) {
                aH[i] = h0[i]; aH[8 + i] = h1[i];
                aL[i] = l0[i]; aL[8 + i] = l1[i];
            }
            if (mt == 0) { aH0 = aH; aL0 = aL; }
            else         { aH1 = aH; aL1 = aL; }
        }

        // ---- B fragments: pre-packed, 32B/lane, coalesced from L2 ----
        size_t boff = ((size_t)(kc * NTILES_N + ntg) * 32 + lane) << 4;
        v16h bch = *(const v16h*)(ch + boff);
        v16h bcl = *(const v16h*)(cl + boff);
        v16h bsh = *(const v16h*)(sh + boff);
        v16h bsl = *(const v16h*)(sl + boff);

        // prefetch next chunk's twiddles (global_prefetch_b8)
        if (kc + 1 < NKC) {
            size_t pnext = ((size_t)((kc + 1) * NTILES_N + ntg) * 32 + lane) << 4;
            __builtin_prefetch(ch + pnext, 0, 1);
            __builtin_prefetch(cl + pnext, 0, 1);
            __builtin_prefetch(sh + pnext, 0, 1);
            __builtin_prefetch(sl + pnext, 0, 1);
        }

        // ---- 12 WMMAs: 2 M-tiles x 3-term split x {cos,sin}, interleaved ----
        accRe0 = __builtin_amdgcn_wmma_f32_16x16x32_f16(false, aH0, false, bch, (short)0, accRe0, false, false);
        accRe1 = __builtin_amdgcn_wmma_f32_16x16x32_f16(false, aH1, false, bch, (short)0, accRe1, false, false);
        accIm0 = __builtin_amdgcn_wmma_f32_16x16x32_f16(false, aH0, false, bsh, (short)0, accIm0, false, false);
        accIm1 = __builtin_amdgcn_wmma_f32_16x16x32_f16(false, aH1, false, bsh, (short)0, accIm1, false, false);
        accRe0 = __builtin_amdgcn_wmma_f32_16x16x32_f16(false, aL0, false, bch, (short)0, accRe0, false, false);
        accRe1 = __builtin_amdgcn_wmma_f32_16x16x32_f16(false, aL1, false, bch, (short)0, accRe1, false, false);
        accIm0 = __builtin_amdgcn_wmma_f32_16x16x32_f16(false, aL0, false, bsh, (short)0, accIm0, false, false);
        accIm1 = __builtin_amdgcn_wmma_f32_16x16x32_f16(false, aL1, false, bsh, (short)0, accIm1, false, false);
        accRe0 = __builtin_amdgcn_wmma_f32_16x16x32_f16(false, aH0, false, bcl, (short)0, accRe0, false, false);
        accRe1 = __builtin_amdgcn_wmma_f32_16x16x32_f16(false, aH1, false, bcl, (short)0, accRe1, false, false);
        accIm0 = __builtin_amdgcn_wmma_f32_16x16x32_f16(false, aH0, false, bsl, (short)0, accIm0, false, false);
        accIm1 = __builtin_amdgcn_wmma_f32_16x16x32_f16(false, aH1, false, bsl, (short)0, accIm1, false, false);

        __syncthreads();
    }

    // ---- epilogue: interleaved complex64 store (float2 per element) ----
    const int n = ntg * 16 + (lane & 15);
    #pragma unroll
    for (int mt = 0; mt < 2; ++mt) {
        const v8f& re = (mt == 0) ? accRe0 : accRe1;
        const v8f& im = (mt == 0) ? accIm0 : accIm1;
        #pragma unroll
        for (int r = 0; r < 8; ++r) {
            int f = f0 + mt * 16 + r + laneHi * 8;   // C/D layout: VGPR r -> M=r / r+8
            if (f < NFRAMES) {
                size_t o = ((size_t)(b * NFRAMES + f) * DFT + n) * 2;
                float2 v;
                v.x = re[r];
                v.y = im[r];
                *(float2*)(out + o) = v;
            }
        }
    }
}

// ---------------------------------------------------------------------------
// Host launcher
// ---------------------------------------------------------------------------
extern "C" void kernel_launch(void* const* d_in, const int* in_sizes, int n_in,
                              void* d_out, int out_size, void* d_ws, size_t ws_size,
                              hipStream_t stream) {
    const float* sig = (const float*)d_in[0];   // (32, 262144) f32
    const float* win = (const float*)d_in[1];   // (1024,) f32
    // d_in[2..4] are the scalar sizes; they match our compile-time constants.

    // workspace: 4 packed twiddle arrays of 1024*1024 halves (2 MB each, 8 MB total)
    const size_t MAT = (size_t)DFT * DFT;
    _Float16* ch = (_Float16*)d_ws;
    _Float16* cl = ch + MAT;
    _Float16* sh = cl + MAT;
    _Float16* sl = sh + MAT;

    dft_pack_kernel<<<(int)(MAT / 256), 256, 0, stream>>>(ch, cl, sh, sl);

    dim3 grid(DFT / BLK_BINS, (NFRAMES + FR_TILE - 1) / FR_TILE, BATCH); // (8, 32, 32)
    stft_wmma_kernel<<<grid, 256, 0, stream>>>(sig, win, ch, cl, sh, sl,
                                               (float*)d_out);
}
